// LFMMIEnergy_82051055223139
// MI455X (gfx1250) — compile-verified, matrix operand order checked by reading, not run on hardware
//
#include <hip/hip_runtime.h>

#define Bn 16
#define Tn 400
#define Vn 3000
#define Sn 10000
#define En 100000
#define SBn (Sn * Bn)
#define NEGF (-1e30f)
#define LOG_LEAKY (-2.3025850929940457f)

// ---- monotonic float <-> uint encoding so float-max can use native u32 atomics ----
__device__ __forceinline__ unsigned fenc(float f) {
    unsigned u = __float_as_uint(f);
    return (u & 0x80000000u) ? ~u : (u | 0x80000000u);
}
__device__ __forceinline__ float fdec(unsigned k) {
    unsigned u = (k & 0x80000000u) ? (k & 0x7fffffffu) : ~k;
    return __uint_as_float(u);
}

// ---- global log-sum-exp of init/final logits (single workgroup) ----
__global__ void __launch_bounds__(1024) kLse(const float* __restrict__ ia,
                                             const float* __restrict__ fa,
                                             float* __restrict__ scal) {
    __shared__ float red[1024];
    int tid = threadIdx.x;
    const float* srcs[2] = {ia, fa};
    for (int w = 0; w < 2; ++w) {
        const float* p = srcs[w];
        float mv = NEGF;
        for (int i = tid; i < Sn; i += 1024) mv = fmaxf(mv, p[i]);
        red[tid] = mv; __syncthreads();
        for (int off = 512; off >= 1; off >>= 1) {
            if (tid < off) red[tid] = fmaxf(red[tid], red[tid + off]);
            __syncthreads();
        }
        float gm = red[0]; __syncthreads();
        float sv = 0.f;
        for (int i = tid; i < Sn; i += 1024) sv += __expf(p[i] - gm);
        red[tid] = sv; __syncthreads();
        for (int off = 512; off >= 1; off >>= 1) {
            if (tid < off) red[tid] += red[tid + off];
            __syncthreads();
        }
        if (tid == 0) scal[w] = gm + logf(red[0]);
        __syncthreads();
    }
}

// ---- initialize alpha, segment-max keys, segment sums, normalized logs ----
__global__ void kInit(const float* __restrict__ init_logits,
                      const float* __restrict__ final_logits,
                      const float* __restrict__ scal,
                      float* __restrict__ alpha, unsigned* __restrict__ mkey,
                      float* __restrict__ sacc, float* __restrict__ init_log,
                      float* __restrict__ final_log) {
    int idx = blockIdx.x * blockDim.x + threadIdx.x;
    if (idx >= SBn) return;
    int s = idx >> 4;
    float il = init_logits[s] - scal[0];
    alpha[idx] = il;
    mkey[idx]  = fenc(NEGF);
    sacc[idx]  = 0.f;
    if ((idx & 15) == 0) {
        init_log[s]  = il;
        final_log[s] = final_logits[s] - scal[1];
    }
}

// ---- pass 1: segmented max over edges (lanes 0..15 share an edge -> coalesced alpha/atomics) ----
__global__ void kEdgeMax(const float* __restrict__ x, const float* __restrict__ trans,
                         const int* __restrict__ in_state, const int* __restrict__ out_state,
                         const int* __restrict__ pdf, const float* __restrict__ alpha,
                         unsigned* __restrict__ mkey, int t) {
    int gid = blockIdx.x * blockDim.x + threadIdx.x;
    if (gid >= En * Bn) return;
    int e = gid >> 4, b = gid & 15;
    float c = alpha[(in_state[e] << 4) + b] + trans[e]
            + x[((long)b * Tn + t) * (long)Vn + pdf[e]];
    atomicMax(&mkey[(out_state[e] << 4) + b], fenc(c));
}

// ---- pass 2: segmented sum of exp(c - max_safe) ----
__global__ void kEdgeSum(const float* __restrict__ x, const float* __restrict__ trans,
                         const int* __restrict__ in_state, const int* __restrict__ out_state,
                         const int* __restrict__ pdf, const float* __restrict__ alpha,
                         const unsigned* __restrict__ mkey, float* __restrict__ sacc, int t) {
    int gid = blockIdx.x * blockDim.x + threadIdx.x;
    if (gid >= En * Bn) return;
    int e = gid >> 4, b = gid & 15;
    int os = out_state[e];
    float c = alpha[(in_state[e] << 4) + b] + trans[e]
            + x[((long)b * Tn + t) * (long)Vn + pdf[e]];
    float m = fdec(mkey[(os << 4) + b]);
    float ms = (m > NEGF * 0.5f) ? m : 0.f;
    atomicAdd(&sacc[(os << 4) + b], __expf(c - ms));
}

// ---- pass 3 (single WGP workgroup, 32 waves): finalize a, leaky redistribution via WMMA ----
__global__ void __launch_bounds__(1024) kFinalize(unsigned* __restrict__ mkey,
                                                  float* __restrict__ sacc,
                                                  float* __restrict__ abuf,
                                                  float* __restrict__ alpha,
                                                  const float* __restrict__ init_log) {
    __shared__ float red[1024];
    __shared__ float maxb_s[16];
    __shared__ float tsum_s[16];
    __shared__ float lk_s[16];
    int tid = threadIdx.x;

    // pass 0: a = m_safe + log(s) (or NEG); re-arm m/s for next timestep
    for (int idx = tid; idx < SBn; idx += 1024) {
        float m  = fdec(mkey[idx]);
        float ss = sacc[idx];
        float ms = (m > NEGF * 0.5f) ? m : 0.f;
        abuf[idx] = (ss > 0.f) ? (ms + logf(fmaxf(ss, 1e-37f))) : NEGF;
        mkey[idx] = fenc(NEGF);
        sacc[idx] = 0.f;
    }
    __threadfence();
    __syncthreads();

    // pass A: per-batch max over states (tid%16 == batch invariant under all strides)
    float mv = NEGF;
    for (int idx = tid; idx < SBn; idx += 1024) mv = fmaxf(mv, abuf[idx]);
    red[tid] = mv; __syncthreads();
    for (int off = 512; off >= 16; off >>= 1) {
        if (tid < off) red[tid] = fmaxf(red[tid], red[tid + off]);
        __syncthreads();
    }
    if (tid < 16) {
        float m = red[tid];
        maxb_s[tid] = (m > NEGF * 0.5f) ? m : 0.f;
        tsum_s[tid] = 0.f;
    }
    __syncthreads();

    // pass B: tot[b] = sum_s exp(a - max[b]) via matrix pipe: D = ones(16x4) * B(4x16) + C
    int lane = tid & 31, wave = tid >> 5;
#if __has_builtin(__builtin_amdgcn_wmma_f32_16x16x4_f32)
    typedef float v2f_t __attribute__((ext_vector_type(2)));
    typedef float v8f_t __attribute__((ext_vector_type(8)));
    v8f_t acc = {0.f, 0.f, 0.f, 0.f, 0.f, 0.f, 0.f, 0.f};
    v2f_t ones; ones.x = 1.f; ones.y = 1.f;
    int bb = lane & 15;                 // B-matrix column N == batch == lane%16
    int kh = (lane >> 4) << 1;          // lanes 0-15 carry K={0,1}, lanes 16-31 K={2,3}
    float mb = maxb_s[bb];
    for (int chunk = wave; chunk < Sn / 4; chunk += 32) {
        int base = chunk << 2;
        v2f_t bv;
        bv.x = __expf(abuf[((base + kh)     << 4) + bb] - mb);
        bv.y = __expf(abuf[((base + kh + 1) << 4) + bb] - mb);
        acc = __builtin_amdgcn_wmma_f32_16x16x4_f32(false, ones, false, bv,
                                                    (short)0, acc, false, false);
    }
    if (lane < 16) atomicAdd(&tsum_s[lane], acc[0]);  // D row 0: lane n -> batch n
#else
    typedef _Float16 v16h_t __attribute__((ext_vector_type(16)));
    typedef float    v8f_t  __attribute__((ext_vector_type(8)));
    v8f_t acc = {0.f, 0.f, 0.f, 0.f, 0.f, 0.f, 0.f, 0.f};
    v16h_t ones;
#pragma unroll
    for (int j = 0; j < 16; ++j) ones[j] = (_Float16)1.0f;
    for (int chunk = wave; chunk < (Sn + 31) / 32; chunk += 32) {
        int st = (chunk << 5) + lane;   // lane%16 -> B-matrix column N, lane/16 -> K half (ones-A)
        v16h_t bv;
#pragma unroll
        for (int j = 0; j < 16; ++j) {  // element j -> B-matrix row K (holds batch j)
            float v = (st < Sn) ? abuf[(st << 4) + j] : NEGF;
            bv[j] = (_Float16)__expf(v - maxb_s[j]);
        }
        acc = __builtin_amdgcn_wmma_f32_16x16x32_f16(false, ones, false, bv,
                                                     (short)0, acc, false, false);
    }
    if (lane < 16) atomicAdd(&tsum_s[lane], acc[0]);
#endif
    __syncthreads();
    if (tid < 16) {
        float tot = maxb_s[tid] + logf(fmaxf(tsum_s[tid], 1e-37f));
        lk_s[tid] = LOG_LEAKY + tot;
    }
    __syncthreads();

    // pass C: alpha = logaddexp(a, log_leaky + init_log + tot)
    for (int idx = tid; idx < SBn; idx += 1024) {
        float u = abuf[idx];
        float v = lk_s[idx & 15] + init_log[idx >> 4];
        float hi = fmaxf(u, v), lo = fminf(u, v);
        alpha[idx] = hi + log1pf(__expf(lo - hi));
    }
}

// ---- epilogue: energy = -logsumexp(alpha + final_log)/T ----
__global__ void __launch_bounds__(1024) kFinal(const float* __restrict__ alpha,
                                               const float* __restrict__ final_log,
                                               float* __restrict__ out) {
    __shared__ float red[1024];
    __shared__ float maxb_s[16];
    int tid = threadIdx.x;
    float mv = NEGF;
    for (int idx = tid; idx < SBn; idx += 1024)
        mv = fmaxf(mv, alpha[idx] + final_log[idx >> 4]);
    red[tid] = mv; __syncthreads();
    for (int off = 512; off >= 16; off >>= 1) {
        if (tid < off) red[tid] = fmaxf(red[tid], red[tid + off]);
        __syncthreads();
    }
    if (tid < 16) maxb_s[tid] = (red[tid] > NEGF * 0.5f) ? red[tid] : 0.f;
    __syncthreads();
    float sv = 0.f;
    for (int idx = tid; idx < SBn; idx += 1024)
        sv += __expf(alpha[idx] + final_log[idx >> 4] - maxb_s[idx & 15]);
    red[tid] = sv; __syncthreads();
    for (int off = 512; off >= 16; off >>= 1) {
        if (tid < off) red[tid] += red[tid + off];
        __syncthreads();
    }
    if (tid < 16) {
        float lz = maxb_s[tid] + logf(fmaxf(red[tid], 1e-37f));
        out[tid] = -lz / (float)Tn;
    }
}

extern "C" void kernel_launch(void* const* d_in, const int* in_sizes, int n_in,
                              void* d_out, int out_size, void* d_ws, size_t ws_size,
                              hipStream_t stream) {
    const float* x            = (const float*)d_in[0];
    const float* trans        = (const float*)d_in[1];
    const float* init_logits  = (const float*)d_in[2];
    const float* final_logits = (const float*)d_in[3];
    const int*   in_state     = (const int*)d_in[4];
    const int*   out_state    = (const int*)d_in[5];
    const int*   pdf          = (const int*)d_in[6];
    float*       out          = (float*)d_out;

    // workspace layout (~2.64 MB, all L2-resident)
    float*    alpha     = (float*)d_ws;
    unsigned* mkey      = (unsigned*)(alpha + SBn);
    float*    sacc      = (float*)(mkey + SBn);
    float*    abuf      = sacc + SBn;
    float*    init_log  = abuf + SBn;
    float*    final_log = init_log + Sn;
    float*    scal      = final_log + Sn;

    kLse<<<1, 1024, 0, stream>>>(init_logits, final_logits, scal);
    kInit<<<(SBn + 255) / 256, 256, 0, stream>>>(init_logits, final_logits, scal,
                                                 alpha, mkey, sacc, init_log, final_log);
    const int egrid = (En * Bn + 255) / 256;
    for (int t = 0; t < Tn; ++t) {
        kEdgeMax<<<egrid, 256, 0, stream>>>(x, trans, in_state, out_state, pdf, alpha, mkey, t);
        kEdgeSum<<<egrid, 256, 0, stream>>>(x, trans, in_state, out_state, pdf, alpha, mkey, sacc, t);
        kFinalize<<<1, 1024, 0, stream>>>(mkey, sacc, abuf, alpha, init_log);
    }
    kFinal<<<1, 1024, 0, stream>>>(alpha, final_log, out);
}
